// FCABlockS2_9234179686828
// MI455X (gfx1250) — matrix operand order repaired and checked by reading, not compiled, not run on hardware
//
#include <hip/hip_runtime.h>
#include <hip/hip_bf16.h>

typedef _Float16 f16;
typedef __attribute__((ext_vector_type(16))) _Float16 hv16;
typedef __attribute__((ext_vector_type(8)))  _Float16 hv8;
typedef __attribute__((ext_vector_type(8)))  float    fv8;

namespace {
constexpr int kNLat = 91, kNLon = 180, kDim = 256, kHeads = 8, kDH = 64, kDK = 192;
constexpr int kHDK = kHeads * kDK;    // 1536
constexpr int kHDH = kHeads * kDH;    // 512
constexpr int kM0  = kNLat * kNLon;   // 16380
constexpr int kMp  = 16384;           // padded rows for WMMA
constexpr int kH1  = 4 * kDim;        // 1024
constexpr int kUC  = kHDH + kDim;     // 768
constexpr int kJLon = 192;            // padded 180
constexpr int kJLat = 96;             // padded 91
constexpr int kN1  = kNLon * kDH;     // 11520
constexpr int kN2  = kNLat * kDH;     // 5824
constexpr int kN2p = 96 * kDH;        // 6144 (rows of phi1T)
constexpr float kPI = 3.14159265358979323846f;
}

__device__ __forceinline__ float gelu_tanh(float x) {
  float x3 = x * x * x;
  return 0.5f * x * (1.0f + tanhf(0.7978845608028654f * (x + 0.044715f * x3)));
}

// ---------------- group-norm stats over u_src (32 groups of 8 ch) -----------
__global__ void k_gn_stats_src(const float* __restrict__ x, float* __restrict__ stats) {
  int g = blockIdx.x, t = threadIdx.x;
  float s = 0.f, ss = 0.f;
  for (int idx = t; idx < kM0 * 8; idx += 256) {
    int r = idx >> 3, c = idx & 7;
    float v = x[(long)r * kDim + g * 8 + c];
    s += v; ss += v * v;
  }
  __shared__ float sh0[256], sh1[256];
  sh0[t] = s; sh1[t] = ss; __syncthreads();
  for (int o = 128; o > 0; o >>= 1) {
    if (t < o) { sh0[t] += sh0[t + o]; sh1[t] += sh1[t + o]; }
    __syncthreads();
  }
  if (t == 0) {
    float n = (float)(kM0 * 8);
    float m = sh0[0] / n;
    float var = sh1[0] / n - m * m;
    stats[g * 2] = m; stats[g * 2 + 1] = rsqrtf(var + 1e-5f);
  }
}

// ---------------- normalize u_src -> f16, zero-pad rows to kMp --------------
__global__ void k_norm_src(const float* __restrict__ x, const float* __restrict__ stats,
                           const float* __restrict__ w, const float* __restrict__ b,
                           f16* __restrict__ out) {
  long idx = (long)blockIdx.x * 256 + threadIdx.x;  // kMp*kDim total
  int c = (int)(idx % kDim);
  long r = idx / kDim;
  float v = 0.f;
  if (r < kM0) {
    int g = c >> 3;
    v = (x[idx] - stats[g * 2]) * stats[g * 2 + 1] * w[c] + b[c];
  }
  out[idx] = (f16)v;
}

// ---------------- transpose-convert weights f32(K,N) -> f16(N,K) ------------
__global__ void k_tconv(const float* __restrict__ in, f16* __restrict__ out, int K, int N) {
  long idx = (long)blockIdx.x * 256 + threadIdx.x;
  if (idx >= (long)K * N) return;
  int k = (int)(idx % K);
  int n = (int)(idx / K);
  out[idx] = (f16)in[(long)k * N + n];
}

__global__ void k_zero32(unsigned int* __restrict__ p, long words) {
  long idx = (long)blockIdx.x * 256 + threadIdx.x;
  if (idx < words) p[idx] = 0u;
}

// ---------------- generic wave32 WMMA GEMM: C = act(A @ B^T + bias) ---------
// One wave computes a 32x64 register tile: 2 M-frags x 4 N-frags -> 8 WMMA
// accumulators per K-step (12 b128 loads : 8 v_wmma).
// A: f16 (rows x K) row-major.  BT: f16 (N x K) row-major.
// mode 0: f32 out + bias; 1: f32 gelu; 2: f16 out; 3: f16 gelu;
// mode 4: f16 scatter  dest[(m*64+(n&63))*sld + (n>>6)]  (attention repack)
__global__ void k_gemm16(const f16* __restrict__ A, int lda, long sA,
                         const f16* __restrict__ BT, int ldb, long sB,
                         const float* __restrict__ bias,
                         void* __restrict__ C, int ldc, long sC,
                         int K, int Mvalid, int mode, int sld) {
  int l = threadIdx.x;                       // 32 lanes, EXEC all-ones
  int half = l >> 4, ln = l & 15;
  int z = blockIdx.z;
  const f16* Ab = A + (long)z * sA;
  const f16* Bb = BT + (long)z * sB;
  const f16* ap0 = Ab + ((long)blockIdx.y * 32 + ln) * lda;
  const f16* ap1 = ap0 + (long)16 * lda;
  const f16* bp0 = Bb + ((long)blockIdx.x * 64 + ln) * ldb;
  const f16* bp1 = bp0 + (long)16 * ldb;
  const f16* bp2 = bp0 + (long)32 * ldb;
  const f16* bp3 = bp0 + (long)48 * ldb;
  fv8 acc[2][4] = {};
  for (int k0 = 0; k0 < K; k0 += 32) {
    union { hv16 v; hv8 h[2]; } a0, a1, b0, b1, b2, b3;
    a0.h[0] = *(const hv8*)(ap0 + k0 + half * 16);
    a0.h[1] = *(const hv8*)(ap0 + k0 + half * 16 + 8);
    a1.h[0] = *(const hv8*)(ap1 + k0 + half * 16);
    a1.h[1] = *(const hv8*)(ap1 + k0 + half * 16 + 8);
    b0.h[0] = *(const hv8*)(bp0 + k0 + half * 16);
    b0.h[1] = *(const hv8*)(bp0 + k0 + half * 16 + 8);
    b1.h[0] = *(const hv8*)(bp1 + k0 + half * 16);
    b1.h[1] = *(const hv8*)(bp1 + k0 + half * 16 + 8);
    b2.h[0] = *(const hv8*)(bp2 + k0 + half * 16);
    b2.h[1] = *(const hv8*)(bp2 + k0 + half * 16 + 8);
    b3.h[0] = *(const hv8*)(bp3 + k0 + half * 16);
    b3.h[1] = *(const hv8*)(bp3 + k0 + half * 16 + 8);
    __builtin_prefetch(ap0 + k0 + 64, 0, 1);
    __builtin_prefetch(bp0 + k0 + 64, 0, 1);
    acc[0][0] = __builtin_amdgcn_wmma_f32_16x16x32_f16(false, a0.v, false, b0.v, (short)0, acc[0][0], false, false);
    acc[1][0] = __builtin_amdgcn_wmma_f32_16x16x32_f16(false, a1.v, false, b0.v, (short)0, acc[1][0], false, false);
    acc[0][1] = __builtin_amdgcn_wmma_f32_16x16x32_f16(false, a0.v, false, b1.v, (short)0, acc[0][1], false, false);
    acc[1][1] = __builtin_amdgcn_wmma_f32_16x16x32_f16(false, a1.v, false, b1.v, (short)0, acc[1][1], false, false);
    acc[0][2] = __builtin_amdgcn_wmma_f32_16x16x32_f16(false, a0.v, false, b2.v, (short)0, acc[0][2], false, false);
    acc[1][2] = __builtin_amdgcn_wmma_f32_16x16x32_f16(false, a1.v, false, b2.v, (short)0, acc[1][2], false, false);
    acc[0][3] = __builtin_amdgcn_wmma_f32_16x16x32_f16(false, a0.v, false, b3.v, (short)0, acc[0][3], false, false);
    acc[1][3] = __builtin_amdgcn_wmma_f32_16x16x32_f16(false, a1.v, false, b3.v, (short)0, acc[1][3], false, false);
  }
#pragma unroll
  for (int mt = 0; mt < 2; ++mt) {
#pragma unroll
    for (int nt = 0; nt < 4; ++nt) {
      int n = blockIdx.x * 64 + nt * 16 + ln;
      float bv = bias ? bias[n] : 0.f;
#pragma unroll
      for (int r = 0; r < 8; ++r) {
        long m = (long)blockIdx.y * 32 + mt * 16 + half * 8 + r;
        if (m >= Mvalid) continue;
        float v = acc[mt][nt][r] + bv;
        if (mode == 1 || mode == 3) v = gelu_tanh(v);
        if (mode <= 1)      ((float*)C)[(long)z * sC + m * ldc + n] = v;
        else if (mode == 4) ((f16*)C)[(long)z * sC + (m * 64 + (n & 63)) * sld + (n >> 6)] = (f16)v;
        else                ((f16*)C)[(long)z * sC + m * ldc + n] = (f16)v;
      }
    }
  }
}

// ---------------- u cols [512,768) -> f16 -----------------------------------
__global__ void k_cvt_umid(const float* __restrict__ u, f16* __restrict__ out) {
  long idx = (long)blockIdx.x * 256 + threadIdx.x;  // kMp*kDim
  long r = idx / kDim; int c = (int)(idx % kDim);
  out[idx] = (f16)u[r * kUC + kHDH + c];
}

// ---------------- pooling reductions ----------------------------------------
__global__ void k_reduce_long(const float* __restrict__ y, float* __restrict__ out) {
  int lon = blockIdx.x, c = threadIdx.x;
  float s = 0.f;
  for (int i = 0; i < kNLat; ++i) s += y[((long)i * kNLon + lon) * kDim + c];
  out[(long)lon * kDim + c] = s * (1.0f / kNLat);
}
__global__ void k_reduce_lat(const float* __restrict__ y, const float* __restrict__ wlat,
                             float* __restrict__ out) {
  int i = blockIdx.x, c = threadIdx.x;
  float s = 0.f;
  for (int lon = 0; lon < kNLon; ++lon) s += y[((long)i * kNLon + lon) * kDim + c];
  out[(long)i * kDim + c] = s * wlat[i] * (1.0f / kNLon);
}

// ---------------- latitude weights ------------------------------------------
__global__ void k_wlat(const float* __restrict__ lat, float* __restrict__ wlat) {
  int i = threadIdx.x;
  if (i >= kNLat) return;
  float w = cosf(lat[i]);
  float dlat = lat[1] - lat[0];
  float sp = sinf(dlat * 0.25f);
  float pw = sp * sp / sinf(dlat * 0.5f);
  if ((i == 0 || i == kNLat - 1) && w < 0.001f) w = pw;   // polar_eps = 1.0
  wlat[i] = w;
}

// ---------------- per-row LayerNorm + gated MLP -----------------------------
__global__ void k_gated_mlp(const float* __restrict__ x,
                            const float* __restrict__ lnw, const float* __restrict__ lnb,
                            const float* __restrict__ w1, const float* __restrict__ b1,
                            const float* __restrict__ w2, const float* __restrict__ b2,
                            float* __restrict__ out) {
  int r = blockIdx.x, t = threadIdx.x;   // 256 threads
  __shared__ float xs[kDim], gs[kDim], red[256];
  float v = x[(long)r * kDim + t];
  red[t] = v; __syncthreads();
  for (int o = 128; o > 0; o >>= 1) { if (t < o) red[t] += red[t + o]; __syncthreads(); }
  float m = red[0] * (1.0f / kDim); __syncthreads();
  float dv = v - m;
  red[t] = dv * dv; __syncthreads();
  for (int o = 128; o > 0; o >>= 1) { if (t < o) red[t] += red[t + o]; __syncthreads(); }
  float var = red[0] * (1.0f / kDim);
  xs[t] = dv * rsqrtf(var + 1e-5f) * lnw[t] + lnb[t];
  __syncthreads();
  float h0 = b1[t], h1 = b1[t + kDim];
  for (int c = 0; c < kDim; ++c) {
    float xv = xs[c];
    h0 += xv * w1[(long)c * (2 * kDim) + t];
    h1 += xv * w1[(long)c * (2 * kDim) + t + kDim];
  }
  gs[t] = gelu_tanh(h0) * h1;
  __syncthreads();
  float o = b2[t];
  for (int c = 0; c < kDim; ++c) o += gs[c] * w2[(long)c * kDim + t];
  out[(long)r * kDim + t] = o;
}

// ---------------- f32(rows,cols) -> f16(rowsPad,cols), zero pad -------------
__global__ void k_cvt_pad16(const float* __restrict__ in, f16* __restrict__ out,
                            int rows, int rowsPad, int cols) {
  long idx = (long)blockIdx.x * 256 + threadIdx.x;
  if (idx >= (long)rowsPad * cols) return;
  long r = idx / cols;
  out[idx] = (r < rows) ? (f16)in[idx] : (f16)0.f;
}

// ---------------- fused Bessel basis + mod-GEMM + low-rank kernel -----------
// For query row i and a tile of 16 source cols j: basis(16xNK) in LDS,
// mod = basis @ W^T via WMMA (never materialized), then
// K[h,i,j] = leaky( sum_n mod[j,n]*q[i,n]*k[j,n] ) * wlat[j] * scale  -> f16 A op.
__global__ void k_fused_lowrank(const float* __restrict__ q,    // NI x 1536
                                const f16*  __restrict__ k16,   // Jpad x 1536 (pad rows zero)
                                const f16*  __restrict__ WT16,  // 1536 x NK
                                const float* __restrict__ freqs,
                                const float* __restrict__ qc,   // query coords (NI)
                                const float* __restrict__ sc,   // source coords (NJ)
                                const float* __restrict__ wlat, // or null
                                f16* __restrict__ outA,         // HEADS x Ipad x Jpad
                                int NI, int NJ, int Ipad, int Jpad,
                                int NK, int periodic, float scale) {
  int i = blockIdx.y;
  int j0 = blockIdx.x * 16;
  int t = threadIdx.x;  // 256 = 8 waves
  __shared__ f16 basis[16][64];
  __shared__ float qs[kHDK];
  __shared__ f16 ks[16][kHDK];
  __shared__ float acc[16][kHeads];

  for (int c = t; c < kHDK; c += 256) qs[c] = q[(long)i * kHDK + c];
  for (int idx = t; idx < 16 * kHDK; idx += 256) {
    int jj = idx / kHDK, c = idx % kHDK;
    ks[jj][c] = k16[(long)(j0 + jj) * kHDK + c];
  }
  float qcoord = qc[i];
  for (int idx = t; idx < 16 * 64; idx += 256) {
    int jj = idx / 64, f = idx % 64;
    int j = j0 + jj;
    float bval = 0.f;
    if (j < NJ && f < NK) {
      float dist = fabsf(sc[j] - qcoord);
      if (periodic && dist > kPI) dist = 2.f * kPI - dist;
      float d = dist + 1e-5f;
      bval = 0.7978845608028654f * sinf(freqs[f] * d / kPI) / d;
    }
    basis[jj][f] = (f16)bval;
  }
  if (t < 16 * kHeads) acc[t >> 3][t & 7] = 0.f;
  __syncthreads();

  int wave = t >> 5, l = t & 31;
  int half = l >> 4, ln = l & 15;
  union { hv16 v; f16 e[16]; } a0, a1;
#pragma unroll
  for (int e = 0; e < 8; ++e) {
    a0.e[e]     = basis[ln][half * 16 + e];
    a0.e[e + 8] = basis[ln][half * 16 + 8 + e];
    a1.e[e]     = basis[ln][32 + half * 16 + e];
    a1.e[e + 8] = basis[ln][32 + half * 16 + 8 + e];
  }
  for (int tt = 0; tt < 12; ++tt) {     // 96 n-tiles / 8 waves
    int n = (wave * 12 + tt) * 16 + ln;
    union { hv16 v; hv8 h[2]; } b0;
    b0.h[0] = *(const hv8*)(WT16 + (long)n * NK + half * 16);
    b0.h[1] = *(const hv8*)(WT16 + (long)n * NK + half * 16 + 8);
    fv8 c = {};
    c = __builtin_amdgcn_wmma_f32_16x16x32_f16(false, a0.v, false, b0.v,
                                               (short)0, c, false, false);
    if (NK == 64) {
      union { hv16 v; hv8 h[2]; } b1;
      b1.h[0] = *(const hv8*)(WT16 + (long)n * NK + 32 + half * 16);
      b1.h[1] = *(const hv8*)(WT16 + (long)n * NK + 32 + half * 16 + 8);
      c = __builtin_amdgcn_wmma_f32_16x16x32_f16(false, a1.v, false, b1.v,
                                                 (short)0, c, false, false);
    }
    int head = n / kDK;
    float qv = qs[n];
#pragma unroll
    for (int r = 0; r < 8; ++r) {
      int jj = half * 8 + r;
      atomicAdd(&acc[jj][head], c[r] * qv * (float)ks[jj][n]);
    }
  }
  __syncthreads();
  if (t < 16 * kHeads) {
    int jj = t >> 3, h = t & 7;
    int j = j0 + jj;
    if (j < NJ) {
      float Kv = acc[jj][h];
      Kv = (Kv >= 0.f) ? Kv : 0.2f * Kv;                     // leaky
      float wj = wlat ? wlat[j] : 1.f;
      outA[((long)h * Ipad + i) * Jpad + j] = (f16)(Kv * wj * scale);
    }
  }
}

// ---------------- v_src -> B^T fragments for attention GEMM1 ----------------
// vT[h][m*64+c][jpad=96] = u[(j*180+m)*768 + h*64 + c], zero for j>=91
__global__ void k_vt_repack(const float* __restrict__ u, f16* __restrict__ vT) {
  long idx = (long)blockIdx.x * 256 + threadIdx.x;  // 8*180*64*96
  int j = (int)(idx % kJLat);
  long r = idx / kJLat;
  int c = (int)(r % kDH); r /= kDH;
  int m = (int)(r % kNLon);
  int h = (int)(r / kNLon);
  float v = 0.f;
  if (j < kNLat) v = u[((long)j * kNLon + m) * kUC + h * kDH + c];
  vT[idx] = (f16)v;
}

// ---------------- group-norm stats over u_phi2 (per head) -------------------
__global__ void k_gn_stats_phi(const float* __restrict__ phi, float* __restrict__ stats) {
  int h = blockIdx.x, t = threadIdx.x;
  const float* p = phi + (long)h * kJLon * kN2;
  float s = 0.f, ss = 0.f;
  for (long idx = t; idx < (long)kNLon * kN2; idx += 256) {
    float v = p[(idx / kN2) * kN2 + idx % kN2];
    s += v; ss += v * v;
  }
  __shared__ float sh0[256], sh1[256];
  sh0[t] = s; sh1[t] = ss; __syncthreads();
  for (int o = 128; o > 0; o >>= 1) {
    if (t < o) { sh0[t] += sh0[t + o]; sh1[t] += sh1[t + o]; }
    __syncthreads();
  }
  if (t == 0) {
    float n = (float)((long)kNLon * kN2);
    float m = sh0[0] / n;
    float var = sh1[0] / n - m * m;
    stats[h * 2] = m; stats[h * 2 + 1] = rsqrtf(var + 1e-5f);
  }
}

// ---------------- normalize+repack u_phi2 -> f16 A (kMp x 512) --------------
__global__ void k_final_repack(const float* __restrict__ phi, const float* __restrict__ stats,
                               const float* __restrict__ gw, const float* __restrict__ gb,
                               f16* __restrict__ A) {
  long idx = (long)blockIdx.x * 256 + threadIdx.x;  // kMp*512
  int ch = (int)(idx % kHDH);
  long r = idx / kHDH;
  float v = 0.f;
  if (r < kM0) {
    int i = (int)(r / kNLon), l = (int)(r % kNLon);
    int h = ch / kDH, c = ch % kDH;
    float x = phi[((long)h * kJLon + l) * kN2 + i * kDH + c];
    v = (x - stats[h * 2]) * stats[h * 2 + 1] * gw[ch] + gb[ch];
  }
  A[idx] = (f16)v;
}

// ============================================================================
extern "C" void kernel_launch(void* const* d_in, const int* in_sizes, int n_in,
                              void* d_out, int out_size, void* d_ws, size_t ws_size,
                              hipStream_t stream) {
  (void)in_sizes; (void)n_in; (void)out_size; (void)ws_size;
  // inputs (setup_inputs dict order)
  const float* u_src      = (const float*)d_in[0];
  const float* u_lat_qry  = (const float*)d_in[1];
  const float* u_long_qry = (const float*)d_in[2];
  const float* src_lat    = (const float*)d_in[3];
  const float* qry_lat    = (const float*)d_in[4];
  const float* src_long   = (const float*)d_in[5];
  const float* qry_long   = (const float*)d_in[6];
  const float* cm_gn_w = (const float*)d_in[7],  *cm_gn_b = (const float*)d_in[8];
  const float* cm_w1 = (const float*)d_in[9],    *cm_b1 = (const float*)d_in[10];
  const float* cm_w2 = (const float*)d_in[11],   *cm_b2 = (const float*)d_in[12];
  const float* tl_in_w = (const float*)d_in[13];
  const float* tl_ln_w = (const float*)d_in[14], *tl_ln_b = (const float*)d_in[15];
  const float* tl_w1 = (const float*)d_in[16],   *tl_b1 = (const float*)d_in[17];
  const float* tl_w2 = (const float*)d_in[18],   *tl_b2 = (const float*)d_in[19];
  const float* ta_in_w = (const float*)d_in[20];
  const float* ta_ln_w = (const float*)d_in[21], *ta_ln_b = (const float*)d_in[22];
  const float* ta_w1 = (const float*)d_in[23],   *ta_b1 = (const float*)d_in[24];
  const float* ta_w2 = (const float*)d_in[25],   *ta_b2 = (const float*)d_in[26];
  const float* klong_wk = (const float*)d_in[27], *klat_wk = (const float*)d_in[28];
  const float* peL_freq = (const float*)d_in[29], *peL_w = (const float*)d_in[30];
  const float* peA_freq = (const float*)d_in[31], *peA_w = (const float*)d_in[32];
  const float* mh_gn_w = (const float*)d_in[33], *mh_gn_b = (const float*)d_in[34];
  const float* mh_w = (const float*)d_in[35];
  float* out = (float*)d_out;

  // ---- workspace layout (needs ~117 MB) ----
  char* w = (char*)d_ws;
  size_t off = 0;
  auto alloc = [&](size_t bytes) -> char* {
    char* p = w + off; off = (off + bytes + 255) & ~(size_t)255; return p;
  };
  float* gn_stats = (float*)alloc(64 * 4);
  float* mh_stats = (float*)alloc(16 * 4);
  float* wlat     = (float*)alloc(kNLat * 4);
  float* uls  = (float*)alloc((size_t)kNLon * kDim * 4);
  float* ula  = (float*)alloc((size_t)kNLat * kDim * 4);
  float* ulsm = (float*)alloc((size_t)kNLon * kDim * 4);
  float* ulam = (float*)alloc((size_t)kNLat * kDim * 4);
  f16* uls16 = (f16*)alloc((size_t)kJLon * kDim * 2);
  f16* ula16 = (f16*)alloc((size_t)kJLat * kDim * 2);
  f16* k16L  = (f16*)alloc((size_t)kJLon * kHDK * 2);
  f16* k16A  = (f16*)alloc((size_t)kJLat * kHDK * 2);
  f16* kAL   = (f16*)alloc((size_t)kHeads * kJLon * kJLon * 2);
  f16* kAA   = (f16*)alloc((size_t)kHeads * kJLat * kJLat * 2);
  f16* w1T   = (f16*)alloc((size_t)kH1 * kDim * 2);
  f16* w2T   = (f16*)alloc((size_t)kUC * kH1 * 2);
  f16* tlT   = (f16*)alloc((size_t)kDim * kDim * 2);
  f16* taT   = (f16*)alloc((size_t)kDim * kDim * 2);
  f16* klT   = (f16*)alloc((size_t)kHDK * kDim * 2);
  f16* kaT   = (f16*)alloc((size_t)kHDK * kDim * 2);
  f16* mhT   = (f16*)alloc((size_t)kDim * kHDH * 2);
  f16* peLT  = (f16*)alloc((size_t)kHDK * 64 * 2);
  f16* peAT  = (f16*)alloc((size_t)kHDK * 32 * 2);
  // big phase-reused regions
  char* R1 = alloc((size_t)kMp * (kDim + kH1) * 2);     // 40 MB
  char* R2 = alloc((size_t)kMp * kUC * 4);              // 48 MB
  char* R4 = alloc((size_t)kHeads * kN1 * kJLat * 2);   // 17.7 MB
  f16*   Xn16   = (f16*)R1;
  f16*   h16    = (f16*)(R1 + (size_t)kMp * kDim * 2);
  f16*   umid16 = (f16*)R1;
  float* yTL    = (float*)(R1 + (size_t)kMp * kDim * 2);
  float* yTA    = (float*)(R1 + (size_t)kMp * kDim * 2 + (size_t)kMp * kDim * 4);
  f16*   phi1T  = (f16*)R1;
  f16*   A16f   = (f16*)(R1 + (size_t)20 * 1024 * 1024);
  float* u      = (float*)R2;
  float* uphi2  = (float*)R2;
  f16*   vT16   = (f16*)R4;

  auto tconv = [&](const float* in, f16* o, int K, int N) {
    long tot = (long)K * N;
    k_tconv<<<dim3((tot + 255) / 256), dim3(256), 0, stream>>>(in, o, K, N);
  };
  auto zero = [&](void* p, size_t bytes) {
    long words = (long)(bytes / 4);
    k_zero32<<<dim3((words + 255) / 256), dim3(256), 0, stream>>>((unsigned int*)p, words);
  };
  // N must be a multiple of 64, Mrows a multiple of 32 (all call sites are).
  auto gemm = [&](const f16* A, int lda, long sA, const f16* BT, int ldb, long sB,
                  const float* bias, void* C, int ldc, long sC,
                  int K, int N, int Mrows, int Mvalid, int mode, int sld, int batch) {
    k_gemm16<<<dim3(N / 64, Mrows / 32, batch), dim3(32), 0, stream>>>(
        A, lda, sA, BT, ldb, sB, bias, C, ldc, sC, K, Mvalid, mode, sld);
  };

  // ---- setup ----
  k_wlat<<<dim3(1), dim3(128), 0, stream>>>(src_lat, wlat);
  k_gn_stats_src<<<dim3(32), dim3(256), 0, stream>>>(u_src, gn_stats);
  k_norm_src<<<dim3((long)kMp * kDim / 256), dim3(256), 0, stream>>>(
      u_src, gn_stats, cm_gn_w, cm_gn_b, Xn16);
  tconv(cm_w1, w1T, kDim, kH1);
  tconv(cm_w2, w2T, kH1, kUC);
  tconv(tl_in_w, tlT, kDim, kDim);
  tconv(ta_in_w, taT, kDim, kDim);
  tconv(klong_wk, klT, kDim, kHDK);
  tconv(klat_wk, kaT, kDim, kHDK);
  tconv(mh_w, mhT, kHDH, kDim);
  tconv(peL_w, peLT, 64, kHDK);
  tconv(peA_w, peAT, 32, kHDK);
  zero(kAL, (size_t)kHeads * kJLon * kJLon * 2);
  zero(kAA, (size_t)kHeads * kJLat * kJLat * 2);

  // ---- channel MLP: h = gelu(Xn@w1+b1); u = h@w2+b2 ----
  gemm(Xn16, kDim, 0, w1T, kDim, 0, cm_b1, h16, kH1, 0, kDim, kH1, kMp, kMp, 3, 0, 1);
  gemm(h16, kH1, 0, w2T, kH1, 0, cm_b2, u, kUC, 0, kH1, kUC, kMp, kMp, 0, 0, 1);
  k_cvt_umid<<<dim3((long)kMp * kDim / 256), dim3(256), 0, stream>>>(u, umid16);

  // ---- pooling paths ----
  gemm(umid16, kDim, 0, tlT, kDim, 0, nullptr, yTL, kDim, 0, kDim, kDim, kMp, kMp, 0, 0, 1);
  gemm(umid16, kDim, 0, taT, kDim, 0, nullptr, yTA, kDim, 0, kDim, kDim, kMp, kMp, 0, 0, 1);
  k_reduce_long<<<dim3(kNLon), dim3(kDim), 0, stream>>>(yTL, uls);
  k_reduce_lat<<<dim3(kNLat), dim3(kDim), 0, stream>>>(yTA, wlat, ula);
  k_gated_mlp<<<dim3(kNLon), dim3(256), 0, stream>>>(uls, tl_ln_w, tl_ln_b,
                                                     tl_w1, tl_b1, tl_w2, tl_b2, ulsm);
  k_gated_mlp<<<dim3(kNLat), dim3(256), 0, stream>>>(ula, ta_ln_w, ta_ln_b,
                                                     ta_w1, ta_b1, ta_w2, ta_b2, ulam);
  k_cvt_pad16<<<dim3(((long)kJLon * kDim + 255) / 256), dim3(256), 0, stream>>>(
      ulsm, uls16, kNLon, kJLon, kDim);
  k_cvt_pad16<<<dim3(((long)kJLat * kDim + 255) / 256), dim3(256), 0, stream>>>(
      ulam, ula16, kNLat, kJLat, kDim);

  // ---- k projections (f16 out) ----
  gemm(uls16, kDim, 0, klT, kDim, 0, nullptr, k16L, kHDK, 0, kDim, kHDK, kJLon, kJLon, 2, 0, 1);
  gemm(ula16, kDim, 0, kaT, kDim, 0, nullptr, k16A, kHDK, 0, kDim, kHDK, kJLat, kJLat, 2, 0, 1);

  // ---- fused Bessel + low-rank kernels (mod never materialized) ----
  k_fused_lowrank<<<dim3(kJLon / 16, kNLon), dim3(256), 0, stream>>>(
      u_long_qry, k16L, peLT, peL_freq, qry_long, src_long, nullptr, kAL,
      kNLon, kNLon, kJLon, kJLon, 64, 1, 2.f * kPI / kNLon);
  k_fused_lowrank<<<dim3(kJLat / 16, kNLat), dim3(256), 0, stream>>>(
      u_lat_qry, k16A, peAT, peA_freq, qry_lat, src_lat, wlat, kAA,
      kNLat, kNLat, kJLat, kJLat, 32, 0, kPI / kNLat);

  // ---- attention application ----
  k_vt_repack<<<dim3((long)kHeads * kNLon * kDH * kJLat / 256), dim3(256), 0, stream>>>(u, vT16);
  zero(phi1T, (size_t)kHeads * kN2p * kJLon * 2);
  // GEMM1: per head (96 x 11520) = kAA(96x96) @ vT^T, scatter to phi1T (BT for GEMM2)
  gemm(kAA, kJLat, (long)kJLat * kJLat, vT16, kJLat, (long)kN1 * kJLat, nullptr,
       phi1T, 0, (long)kN2p * kJLon, kJLat, kN1, kJLat, kJLat, 4, kJLon, kHeads);
  // GEMM2: per head (192 x 5824) = kAL(192x192) @ phi1T^T -> u_phi2 (f32)
  gemm(kAL, kJLon, (long)kJLon * kJLon, phi1T, kJLon, (long)kN2p * kJLon, nullptr,
       uphi2, kN2, (long)kJLon * kN2, kJLon, kN2, kJLon, kJLon, 0, 0, kHeads);

  // ---- final group-norm + projection ----
  k_gn_stats_phi<<<dim3(kHeads), dim3(256), 0, stream>>>(uphi2, mh_stats);
  k_final_repack<<<dim3((long)kMp * kHDH / 256), dim3(256), 0, stream>>>(
      uphi2, mh_stats, mh_gn_w, mh_gn_b, A16f);
  gemm(A16f, kHDH, 0, mhT, kHDH, 0, nullptr, out, kDim, 0, kHDH, kDim, kMp, kM0, 0, 0, 1);
}
